// GroupedQueryAttention_59201829208398
// MI455X (gfx1250) — matrix operand order repaired and checked by reading, not compiled, not run on hardware
//
#include <hip/hip_runtime.h>

// GQA attention block for MI455X (gfx1250, wave32, WMMA).
// Compute-bound (~310 GFLOP vs ~400MB traffic @ 23.3TB/s) -> all GEMMs via
// v_wmma_f32_16x16x32_bf16 (bf16 in, f32 accumulate). All WMMA fragments are
// assembled with aligned 128-bit loads; GEMM is double-buffered through LDS;
// attention prefetches fragments so loads overlap WMMA/softmax.

#define B_  2
#define S_  2048
#define E_  2048
#define H_  32
#define KV_ 8
#define D_  128
#define GRP (H_ / KV_)
#define EPS 1e-6f

typedef __attribute__((ext_vector_type(16))) __bf16 bf16x16;
typedef __attribute__((ext_vector_type(8)))  __bf16 bf16x8;
typedef __attribute__((ext_vector_type(8)))  float  f32x8;

// ---------------------------------------------------------------------------
// WMMA fragment helpers (contiguous k order for A and B, wave32).
// A-matrix 16x32 bf16: lane m = lane&15, kbase = (lane>>4)*8;
//   elem e -> K = 2*kbase + e = 16*half + e (same order as B).
// B-matrix 32x16 bf16: lane n = lane&15, half = lane>>4;
//   elem e -> K = 16*half + e  -> one contiguous 16-element chunk.
// C/D 16x16 f32: vgpr r -> row m = r + 8*half, col n = lane&15.
// ---------------------------------------------------------------------------
__device__ inline bf16x16 cat16(bf16x8 lo, bf16x8 hi) {
  return __builtin_shufflevector(lo, hi, 0, 1, 2, 3, 4, 5, 6, 7,
                                 8, 9, 10, 11, 12, 13, 14, 15);
}
__device__ inline bf16x16 frag_a(const __bf16* rowp, int kbase) {
  bf16x8 lo = *(const bf16x8*)(rowp + 2 * kbase);
  bf16x8 hi = *(const bf16x8*)(rowp + 2 * kbase + 8);
  return cat16(lo, hi);
}
__device__ inline bf16x16 frag_b16(const __bf16* p) {
  bf16x8 lo = *(const bf16x8*)(p);
  bf16x8 hi = *(const bf16x8*)(p + 8);
  return cat16(lo, hi);
}
__device__ inline f32x8 wmma_bf16(bf16x16 a, bf16x16 b, f32x8 c) {
  return __builtin_amdgcn_wmma_f32_16x16x32_bf16(false, a, false, b, (short)0,
                                                 c, false, false);
}

// ---------------------------------------------------------------------------
// GEMM: Y[M,N] = A[M,K] @ W[K,N] (+ bias). Double-buffered LDS, register-
// staged prefetch: next tile's global loads issue before the WMMA chain,
// LDS stores after it, one barrier per K-step. W tile stored TRANSPOSED so
// B fragments are contiguous. Block = 128 thr (4 waves), tile 64x128, K-step
// 32. Wave w computes rows [16w,16w+16) x 128 cols (8 WMMA per K-step).
// ---------------------------------------------------------------------------
#define TM 64
#define TN 128
#define TK 32
#define SA_STRIDE  40  // bf16; multiple of 8 -> 16B-aligned rows
#define SBT_STRIDE 40

template <int ABF16>
__global__ __launch_bounds__(128) void gemm_bias_kernel(
    const void* __restrict__ Aptr, const float* __restrict__ W,
    const float* __restrict__ bias, float* __restrict__ Y,
    int M, int N, int K) {
  __shared__ __attribute__((aligned(16))) __bf16 sA[2][TM * SA_STRIDE];
  __shared__ __attribute__((aligned(16))) __bf16 sBT[2][TN * SBT_STRIDE];

  const int tid  = threadIdx.x;
  const int lane = tid & 31;
  const int wave = tid >> 5;
  const int m0 = blockIdx.y * TM;
  const int n0 = blockIdx.x * TN;

  const int half  = lane >> 4;
  const int nn    = lane & 15;
  const int kbase = half * 8;

  const int ar  = tid >> 1;         // A staging: row, 16 elems/thread
  const int acb = (tid & 1) * 16;

  auto load_a = [&](int k0, bf16x8& v0, bf16x8& v1) {
    if constexpr (ABF16 != 0) {
      const __bf16* src = (const __bf16*)Aptr + (size_t)(m0 + ar) * K + k0 + acb;
      v0 = *(const bf16x8*)src;
      v1 = *(const bf16x8*)(src + 8);
    } else {
      const float* src = (const float*)Aptr + (size_t)(m0 + ar) * K + k0 + acb;
#pragma unroll
      for (int e = 0; e < 8; ++e) {
        v0[e] = (__bf16)src[e];
        v1[e] = (__bf16)src[e + 8];
      }
    }
  };
  auto load_b = [&](int k0, bf16x8 (&v)[4]) {
    const float* src = W + (size_t)k0 * N + n0 + tid;  // coalesced over tid
#pragma unroll
    for (int e = 0; e < TK; ++e) v[e >> 3][e & 7] = (__bf16)src[(size_t)e * N];
  };
  auto store_tile = [&](int buf, bf16x8 a0, bf16x8 a1, bf16x8 (&bv)[4]) {
    *(bf16x8*)&sA[buf][ar * SA_STRIDE + acb]     = a0;
    *(bf16x8*)&sA[buf][ar * SA_STRIDE + acb + 8] = a1;
#pragma unroll
    for (int j = 0; j < 4; ++j)
      *(bf16x8*)&sBT[buf][tid * SBT_STRIDE + j * 8] = bv[j];
  };

  f32x8 acc[8] = {};
  bf16x8 pa0, pa1, pb[4];

  // prologue: tile 0 -> buffer 0
  load_a(0, pa0, pa1);
  load_b(0, pb);
  store_tile(0, pa0, pa1, pb);
  __syncthreads();

  const int nK = K / TK;
  for (int t = 0; t < nK; ++t) {
    const int buf = t & 1;
    if (t + 1 < nK) {  // issue next tile's global loads (overlap with WMMA)
      load_a((t + 1) * TK, pa0, pa1);
      load_b((t + 1) * TK, pb);
    }
    bf16x16 a = frag_a(&sA[buf][(wave * 16 + nn) * SA_STRIDE], kbase);
#pragma unroll
    for (int c = 0; c < 8; ++c) {
      bf16x16 b = frag_b16(&sBT[buf][(c * 16 + nn) * SBT_STRIDE + 16 * half]);
      acc[c] = wmma_bf16(a, b, acc[c]);
    }
    if (t + 1 < nK) store_tile(buf ^ 1, pa0, pa1, pb);
    __syncthreads();
  }

#pragma unroll
  for (int c = 0; c < 8; ++c) {
    const int col = n0 + c * 16 + nn;
    const float bv = bias ? bias[col] : 0.f;
#pragma unroll
    for (int r = 0; r < 8; ++r) {
      const int row = m0 + wave * 16 + r + 8 * half;
      Y[(size_t)row * N + col] = acc[c][r] + bv;
    }
  }
}

// ---------------------------------------------------------------------------
// RMSNorm + RoPE per (b,s,head): one wave per row of D=128 (4 elems/lane).
// Input  Yin[(b*S+s)*nh + h][d] fp32, output Out[b][h][s][d] bf16.
// outScale folds 1/sqrt(D) into Q.
// ---------------------------------------------------------------------------
__global__ __launch_bounds__(128) void rmsnorm_rope_kernel(
    const float* __restrict__ Yin, __bf16* __restrict__ Out,
    const float* __restrict__ scale, const float* __restrict__ cosT,
    const float* __restrict__ sinT, int nh, float outScale) {
  const int lane = threadIdx.x & 31;
  const int wave = threadIdx.x >> 5;
  const int row  = blockIdx.x * 4 + wave;  // over B*S*nh
  const int h  = row % nh;
  const int bs = row / nh;
  const int s  = bs % S_;
  const int b  = bs / S_;

  const float* src = Yin + (size_t)row * D_;
  const int d0 = lane * 4;
  float x[4];
#pragma unroll
  for (int j = 0; j < 4; ++j) x[j] = src[d0 + j];

  float ss = x[0] * x[0] + x[1] * x[1] + x[2] * x[2] + x[3] * x[3];
#pragma unroll
  for (int off = 16; off > 0; off >>= 1) ss += __shfl_xor(ss, off, 32);
  const float rms = rsqrtf(ss * (1.f / D_) + EPS);

  const float* cr = cosT + (size_t)s * D_ + d0;
  const float* sr = sinT + (size_t)s * D_ + d0;
  __bf16* dst = Out + (((size_t)b * nh + h) * S_ + s) * D_;

  float n[4];
#pragma unroll
  for (int j = 0; j < 4; ++j) n[j] = x[j] * rms * scale[d0 + j];
#pragma unroll
  for (int j = 0; j < 4; ++j) {
    const float partner = __shfl_xor(n[j], 16, 32);  // x[d +/- 64]
    const float rot = (lane < 16) ? -partner : partner;
    dst[d0 + j] = (__bf16)((n[j] * cr[j] + rot * sr[j]) * outScale);
  }
}

// V: fp32 [(b*S+s)*KV + kv][d] -> bf16 TRANSPOSED [b][kv][d][s]
__global__ void vconv_kernel(const float* __restrict__ Vin,
                             __bf16* __restrict__ Vt) {
  const size_t i = (size_t)blockIdx.x * blockDim.x + threadIdx.x;
  const int d = (int)(i % D_);
  size_t t = i / D_;
  const int kv = (int)(t % KV_);
  const size_t bs = t / KV_;
  const int s = (int)(bs % S_);
  const int b = (int)(bs / S_);
  Vt[(((size_t)b * KV_ + kv) * D_ + d) * S_ + s] = (__bf16)Vin[i];
}

// ---------------------------------------------------------------------------
// Causal flash attention: 1 wave per (b,h, 16-query tile), 32-key steps.
// QK^T: 8 WMMAs/step with preloaded K fragments; online softmax via LDS with
// V fragments prefetched underneath it; P@V: 8 WMMAs/step from registers.
// Q already scaled by 1/sqrt(D).
// ---------------------------------------------------------------------------
__global__ __launch_bounds__(32) void attn_kernel(
    const __bf16* __restrict__ Qh, const __bf16* __restrict__ Kh,
    const __bf16* __restrict__ Vt, __bf16* __restrict__ Ctx) {
  __shared__ float sS[16 * 32];
  __shared__ __attribute__((aligned(16))) __bf16 sP[16 * 32];
  __shared__ float sCorr[16];
  __shared__ float sL[16];

  const int lane = threadIdx.x;
  const int qt = blockIdx.x % (S_ / 16);
  const int bh = blockIdx.x / (S_ / 16);
  const int h  = bh % H_;
  const int b  = bh / H_;
  const int hk = h / GRP;

  const int half  = lane >> 4;
  const int nn    = lane & 15;
  const int kbase = half * 8;

  // Q tile: 4 A-fragments covering D=128
  const __bf16* Qrow = Qh + (((size_t)b * H_ + h) * S_ + qt * 16 + nn) * D_;
  bf16x16 aq[4];
#pragma unroll
  for (int c = 0; c < 4; ++c) aq[c] = frag_a(Qrow + 32 * c, kbase);

  f32x8 acc[8] = {};
  float rowm = -1e30f, rowl = 0.f;

  const __bf16* Kbase = Kh + ((size_t)b * KV_ + hk) * S_ * D_;  // [s][d]
  const __bf16* Vtb   = Vt + ((size_t)b * KV_ + hk) * D_ * S_;  // [d][s]

  const int qrow = qt * 16;
  const int nkt  = (qt >> 1) + 1;  // 32-key tiles covering [0, qrow+16)

  for (int kt = 0; kt < nkt; ++kt) {
    __syncthreads();  // protect sS/sP reuse across iterations

    // ---- prefetch this tile's 8 V fragments (consumed after softmax) ----
    bf16x16 bv[8];
#pragma unroll
    for (int c = 0; c < 8; ++c)
      bv[c] = frag_b16(&Vtb[(size_t)(c * 16 + nn) * S_ + kt * 32 + 16 * half]);

    // ---- scores S[16q x 32k]: preload K frags, then WMMA chain ----
#pragma unroll
    for (int ch = 0; ch < 2; ++ch) {
      const __bf16* krow = Kbase + ((size_t)kt * 32 + ch * 16 + nn) * D_;
      bf16x16 bk[4];
#pragma unroll
      for (int c = 0; c < 4; ++c) bk[c] = frag_b16(krow + 32 * c + 16 * half);
      f32x8 sf = {};
#pragma unroll
      for (int c = 0; c < 4; ++c) sf = wmma_bf16(aq[c], bk[c], sf);
#pragma unroll
      for (int r = 0; r < 8; ++r)
        sS[(r + 8 * half) * 32 + ch * 16 + nn] = sf[r];
    }
    __syncthreads();

    // ---- online softmax: lanes 0..15 own one query row each ----
    if (lane < 16) {
      const int q = qrow + lane;
      float mx = rowm;
#pragma unroll
      for (int j = 0; j < 32; ++j) {
        const int kk = kt * 32 + j;
        const float v = (kk <= q) ? sS[lane * 32 + j] : -1e30f;
        sS[lane * 32 + j] = v;
        mx = fmaxf(mx, v);
      }
      float sum = 0.f;
#pragma unroll
      for (int j = 0; j < 32; ++j) {
        const float p = __expf(sS[lane * 32 + j] - mx);
        sP[lane * 32 + j] = (__bf16)p;
        sum += p;
      }
      const float corr = __expf(rowm - mx);
      rowl = rowl * corr + sum;
      rowm = mx;
      sCorr[lane] = corr;
    }
    __syncthreads();

    // ---- rescale accumulators, then acc += P @ V (V already in regs) ----
#pragma unroll
    for (int r = 0; r < 8; ++r) {
      const float f = sCorr[r + 8 * half];
#pragma unroll
      for (int c = 0; c < 8; ++c) acc[c][r] *= f;
    }
    bf16x16 ap = frag_a(&sP[nn * 32], kbase);
#pragma unroll
    for (int c = 0; c < 8; ++c) acc[c] = wmma_bf16(ap, bv[c], acc[c]);
  }

  if (lane < 16) sL[lane] = rowl;
  __syncthreads();

  // ctx[b][s][h*D + d], bf16, s = qrow + (r + 8*half)
  __bf16* out = Ctx + (size_t)(b * S_ + qrow) * (H_ * D_) + h * D_;
#pragma unroll
  for (int r = 0; r < 8; ++r) {
    const float inv = 1.f / sL[r + 8 * half];
#pragma unroll
    for (int c = 0; c < 8; ++c)
      out[(size_t)(r + 8 * half) * (H_ * D_) + c * 16 + nn] =
          (__bf16)(acc[c][r] * inv);
  }
}

// ---------------------------------------------------------------------------
extern "C" void kernel_launch(void* const* d_in, const int* in_sizes, int n_in,
                              void* d_out, int out_size, void* d_ws,
                              size_t ws_size, hipStream_t stream) {
  const float* x    = (const float*)d_in[0];
  // d_in[1]: mask (unused; causal structure handled analytically)
  const float* cosT = (const float*)d_in[2];
  const float* sinT = (const float*)d_in[3];
  const float* Wq   = (const float*)d_in[4];
  const float* bq   = (const float*)d_in[5];
  const float* Wk   = (const float*)d_in[6];
  const float* bk   = (const float*)d_in[7];
  const float* Wv   = (const float*)d_in[8];
  const float* bv   = (const float*)d_in[9];
  const float* Wo   = (const float*)d_in[10];
  const float* qs   = (const float*)d_in[11];
  const float* ks   = (const float*)d_in[12];
  float* out = (float*)d_out;

  char* ws = (char*)d_ws;
  size_t off = 0;
  auto alloc = [&](size_t bytes) {
    char* p = ws + off;
    off = (off + bytes + 255) & ~(size_t)255;
    return p;
  };
  float*  Qlin = (float*)alloc((size_t)B_ * S_ * H_  * D_ * 4);
  float*  Klin = (float*)alloc((size_t)B_ * S_ * KV_ * D_ * 4);
  float*  Vlin = (float*)alloc((size_t)B_ * S_ * KV_ * D_ * 4);
  __bf16* Qh   = (__bf16*)alloc((size_t)B_ * H_  * S_ * D_ * 2);
  __bf16* Kh   = (__bf16*)alloc((size_t)B_ * KV_ * S_ * D_ * 2);
  __bf16* Vt   = (__bf16*)alloc((size_t)B_ * KV_ * S_ * D_ * 2);
  __bf16* Ctx  = (__bf16*)alloc((size_t)B_ * S_ * H_ * D_ * 2);

  const int M = B_ * S_;
  const dim3 blk(128);

  // QKV projections (fp32 A, fp32 W -> bf16 WMMA, fp32 out + bias)
  gemm_bias_kernel<0><<<dim3((H_ * D_) / TN, M / TM), blk, 0, stream>>>(
      x, Wq, bq, Qlin, M, H_ * D_, E_);
  gemm_bias_kernel<0><<<dim3((KV_ * D_) / TN, M / TM), blk, 0, stream>>>(
      x, Wk, bk, Klin, M, KV_ * D_, E_);
  gemm_bias_kernel<0><<<dim3((KV_ * D_) / TN, M / TM), blk, 0, stream>>>(
      x, Wv, bv, Vlin, M, KV_ * D_, E_);

  const float qscale = 0.08838834764831845f;  // 1/sqrt(128), folded into Q
  rmsnorm_rope_kernel<<<(M * H_) / 4, 128, 0, stream>>>(
      Qlin, Qh, qs, cosT, sinT, H_, qscale);
  rmsnorm_rope_kernel<<<(M * KV_) / 4, 128, 0, stream>>>(
      Klin, Kh, ks, cosT, sinT, KV_, 1.0f);
  vconv_kernel<<<(int)(((size_t)B_ * S_ * KV_ * D_) / 256), 256, 0, stream>>>(
      Vlin, Vt);

  attn_kernel<<<B_ * H_ * (S_ / 16), 32, 0, stream>>>(Qh, Kh, Vt, Ctx);

  // out = ctx @ Wo (bf16 A)
  gemm_bias_kernel<1><<<dim3(E_ / TN, M / TM), blk, 0, stream>>>(
      Ctx, Wo, nullptr, out, M, E_, H_ * D_);
}
